// PerformerAttention_71116068487405
// MI455X (gfx1250) — matrix operand order repaired and checked by reading, not compiled, not run on hardware
//
#include <hip/hip_runtime.h>
#include <hip/hip_bf16.h>

// ---------------------------------------------------------------------------
// Types for CDNA5 WMMA (wave32): V_WMMA_F32_16X16X32_BF16
// ---------------------------------------------------------------------------
typedef __bf16 bf16_t;
typedef bf16_t bf16x16 __attribute__((ext_vector_type(16)));
typedef float  floatx8 __attribute__((ext_vector_type(8)));

__device__ __forceinline__ floatx8 wmma_bf16(bf16x16 a, bf16x16 b, floatx8 c) {
    // (neg_a, A, neg_b, B, c_mod, C, reuse_a, reuse_b)
    return __builtin_amdgcn_wmma_f32_16x16x32_bf16(false, a, false, b, (short)0, c,
                                                   false, false);
}

// f32 <-> bf16 bit helpers (round-to-nearest-even)
__device__ __forceinline__ unsigned short f2u(float f) {
    unsigned u = __builtin_bit_cast(unsigned, f);
    unsigned r = u + 0x7fffu + ((u >> 16) & 1u);
    return (unsigned short)(r >> 16);
}
__device__ __forceinline__ float u2f(unsigned short h) {
    return __builtin_bit_cast(float, ((unsigned)h) << 16);
}
__device__ __forceinline__ unsigned pack2(float a, float b) {
    return (unsigned)f2u(a) | ((unsigned)f2u(b) << 16);
}

// ---------------------------------------------------------------------------
// Fragment loads per CDNA5 ISA 7.12.2 layouts (wave32).
// A (16x32 bf16): lanes 0-15 -> M=lane, K {0..7,16..23}; lanes 16-31 -> same
//                 M, K {8..15,24..31}. Two contiguous 16B LDS reads.
// B (32x16 bf16): staged TRANSPOSED in LDS as Bt[N][K]; lane's K-values match
//                 A: K {0..7,16..23} / {8..15,24..31} -> two 16B LDS reads.
// C/D (16x16 f32): VGPR r -> row r + 8*(lane>=16), col = lane&15.
// ---------------------------------------------------------------------------
__device__ __forceinline__ bf16x16
load_frag_a(const unsigned short* Abase, int ld, int m0, int k0, int lane) {
    int mr = m0 + (lane & 15);
    int kh = ((lane >> 4) & 1) * 8;
    union { bf16x16 v; uint4 q[2]; } u;
    const unsigned short* p = Abase + mr * ld + k0 + kh;
    u.q[0] = *(const uint4*)(p);
    u.q[1] = *(const uint4*)(p + 16);
    return u.v;
}

__device__ __forceinline__ bf16x16
load_frag_b(const unsigned short* Btbase, int ldk, int k0, int n0, int lane) {
    int c  = n0 + (lane & 15);
    int kh = ((lane >> 4) & 1) * 8;
    union { bf16x16 v; uint4 q[2]; } u;
    const unsigned short* p = Btbase + c * ldk + k0 + kh;
    u.q[0] = *(const uint4*)(p);
    u.q[1] = *(const uint4*)(p + 16);
    return u.v;
}

// Problem constants
#define BB 4
#define NN 4096
#define CC 1024
#define HH 16
#define DD 64
#define MM 256
#define DATA_NORM 0.35355339059327373f  // 64^-0.25
#define INV_SQRT_M 0.0625f              // 1/sqrt(256)

// ===========================================================================
// Kernel 1: qkv = x @ W_qkv (+bias), scatter into bf16 [B,h,N,d] head layout.
// 128x128 tile per 256-thread workgroup; 8 waves, each 2(M)x4(N) 16x16 tiles.
// ===========================================================================
__global__ __launch_bounds__(256) void qkv_gemm(const float* __restrict__ x,
                                                const float* __restrict__ w,
                                                const float* __restrict__ bias,
                                                unsigned short* __restrict__ qh,
                                                unsigned short* __restrict__ kh,
                                                unsigned short* __restrict__ vh) {
    __shared__ __align__(16) unsigned short As[128 * 64];  // A row-major [128][64]
    __shared__ __align__(16) unsigned short Bt[128 * 64];  // B transposed [n][k]
    const int tid = threadIdx.x, lane = tid & 31, wave = tid >> 5;
    const int m_base = blockIdx.y * 128;
    const int n_base = blockIdx.x * 128;
    const int wm = (wave & 3) * 32;
    const int wn = (wave >> 2) * 64;

    floatx8 acc[2][4] = {};

    for (int k0 = 0; k0 < CC; k0 += 64) {
        __syncthreads();
        // A stage: float4 loads -> packed bf16x4 (uint2) LDS stores
        for (int i = tid; i < 2048; i += 256) {
            int r = i >> 4, c4 = (i & 15) * 4;
            const float4 f = *(const float4*)(x + (size_t)(m_base + r) * CC + k0 + c4);
            uint2 t;
            t.x = pack2(f.x, f.y);
            t.y = pack2(f.z, f.w);
            ((uint2*)As)[i] = t;
        }
        // B stage: coalesced float4 along N, transposed bf16 scatter into Bt[n][k]
        for (int i = tid; i < 2048; i += 256) {
            int kk = i >> 5, n = (i & 31) * 4;
            const float4 f = *(const float4*)(w + (size_t)(k0 + kk) * (3 * CC) + n_base + n);
            Bt[(n + 0) * 64 + kk] = f2u(f.x);
            Bt[(n + 1) * 64 + kk] = f2u(f.y);
            Bt[(n + 2) * 64 + kk] = f2u(f.z);
            Bt[(n + 3) * 64 + kk] = f2u(f.w);
        }
        // Prefetch next K-chunk while WMMA chews on this one (global_prefetch_b8)
        if (k0 + 64 < CC) {
            __builtin_prefetch(x + (size_t)(m_base + (tid >> 1)) * CC + k0 + 64 + (tid & 1) * 32, 0, 0);
            __builtin_prefetch(w + (size_t)(k0 + 64 + (tid >> 2)) * (3 * CC) + n_base + (tid & 3) * 32, 0, 0);
        }
        __syncthreads();
#pragma unroll
        for (int ks = 0; ks < 64; ks += 32) {
            bf16x16 a0 = load_frag_a(As, 64, wm, ks, lane);
            bf16x16 a1 = load_frag_a(As, 64, wm + 16, ks, lane);
#pragma unroll
            for (int j = 0; j < 4; ++j) {
                bf16x16 b = load_frag_b(Bt, 64, ks, wn + j * 16, lane);
                acc[0][j] = wmma_bf16(a0, b, acc[0][j]);
                acc[1][j] = wmma_bf16(a1, b, acc[1][j]);
            }
        }
    }
    // Epilogue: the 128-wide N tile never crosses a 1024 boundary -> q/k/v
    // select is workgroup-uniform (scalar branch, no per-lane exec churn).
    const int which = n_base >> 10;  // 0:q 1:k 2:v
    unsigned short* __restrict__ dst = (which == 0) ? qh : (which == 1) ? kh : vh;
#pragma unroll
    for (int i = 0; i < 2; ++i)
#pragma unroll
        for (int j = 0; j < 4; ++j)
#pragma unroll
            for (int r = 0; r < 8; ++r) {
                int row = m_base + wm + i * 16 + ((lane >> 4) & 1) * 8 + r;
                int col = n_base + wn + j * 16 + (lane & 15);
                float val = acc[i][j][r] + bias[col];
                int hh = (col >> 6) & 15;
                int dd = col & 63;
                int b = row >> 12;
                int n = row & (NN - 1);
                dst[((size_t)(b * HH + hh) * NN + n) * DD + dd] = f2u(val);
            }
}

// ===========================================================================
// Kernel 2: per (b,h): k' = featmap(k); kv += k'^T v ; k_sum += sum_n k'.
// One workgroup per (b,h). 64-row chunks of N. kv[256,64] in persistent regs.
// Feature map parallelized 4 threads/row via LDS partial reductions.
// ===========================================================================
__global__ __launch_bounds__(256) void kv_kernel(const unsigned short* __restrict__ khg,
                                                 const unsigned short* __restrict__ vhg,
                                                 const float* __restrict__ proj,
                                                 unsigned short* __restrict__ kvb,
                                                 float* __restrict__ ksum_g) {
    // LDS carve (~138 KB): projB | kb | kpt | (xp f32 overlapped with vt) | partials
    __shared__ __align__(16) unsigned char smem[141312];
    unsigned short* projB = (unsigned short*)(smem);           // [256][64] (= Bt of proj^T)
    unsigned short* kb    = (unsigned short*)(smem + 32768);   // [64][64]  k chunk (A)
    unsigned short* kpt   = (unsigned short*)(smem + 40960);   // [256][64] k'^T (A of kv gemm)
    float*          xp    = (float*)(smem + 73728);            // [64][256] xproj f32
    unsigned short* vt    = (unsigned short*)(smem + 73728);   // [64][64]  v^T (overlaps xp)
    float*          psq   = (float*)(smem + 139264);           // [64][4] sq partials
    float*          pmx   = (float*)(smem + 140288);           // [64][4] max partials

    const int tid = threadIdx.x, lane = tid & 31, wave = tid >> 5;
    const int bh = blockIdx.x;          // 0..63
    const int h  = bh & (HH - 1);

    // proj[h] f32 -> bf16 (same layout serves as transposed-B for x@proj^T)
    for (int i = tid; i < MM * DD / 4; i += 256) {
        const float4 f = ((const float4*)(proj + (size_t)h * MM * DD))[i];
        uint2 t;
        t.x = pack2(f.x, f.y);
        t.y = pack2(f.z, f.w);
        ((uint2*)projB)[i] = t;
    }

    const unsigned short* kg = khg + (size_t)bh * NN * DD;
    const unsigned short* vg = vhg + (size_t)bh * NN * DD;

    floatx8 kvacc[2][4] = {};
    float ksacc = 0.0f;  // m = tid column of k_sum
    const int fr = tid & 63, fq = tid >> 6;  // feature-map row / quarter

    for (int n0 = 0; n0 < NN; n0 += 64) {
        __syncthreads();
        for (int i = tid; i < 512; i += 256)
            ((uint4*)kb)[i] = *(const uint4*)(kg + (size_t)n0 * DD + i * 8);
        if (tid < 128 && n0 + 64 < NN) {
            const unsigned short* pf = ((tid < 64) ? kg : vg) + (size_t)(n0 + 64) * DD +
                                       (size_t)(tid & 63) * 64;
            __builtin_prefetch(pf, 0, 0);
        }
        __syncthreads();
        // xproj = (k * DN) @ proj^T : M=64, N=256, K=64
        {
            floatx8 xacc[8] = {};
            const int mt = (wave & 3) * 16;
            const int nb = (wave >> 2) * 128;
#pragma unroll
            for (int ks = 0; ks < 64; ks += 32) {
                bf16x16 a = load_frag_a(kb, 64, mt, ks, lane);
#pragma unroll
                for (int j = 0; j < 8; ++j) {
                    bf16x16 b = load_frag_b(projB, 64, ks, nb + j * 16, lane);
                    xacc[j] = wmma_bf16(a, b, xacc[j]);
                }
            }
#pragma unroll
            for (int j = 0; j < 8; ++j)
#pragma unroll
                for (int r = 0; r < 8; ++r) {
                    int row = mt + ((lane >> 4) & 1) * 8 + r;
                    int col = nb + j * 16 + (lane & 15);
                    xp[row * MM + col] = xacc[j][r] * DATA_NORM;
                }
        }
        __syncthreads();
        // Feature map, 4 threads per row. Phase A: sq-norm partials.
        {
            float s = 0.0f;
            for (int dd = fq * 16; dd < fq * 16 + 16; ++dd) {
                float v = u2f(kb[fr * DD + dd]) * DATA_NORM;
                s += v * v;
            }
            psq[fr * 4 + fq] = s;
        }
        __syncthreads();
        // Phase B: subtract sq, rowwise-max partials over 64 columns each.
        {
            float sq = 0.5f * (psq[fr * 4 + 0] + psq[fr * 4 + 1] +
                               psq[fr * 4 + 2] + psq[fr * 4 + 3]);
            float mx = -1e30f;
            for (int m = fq * 64; m < fq * 64 + 64; ++m) {
                float v = xp[fr * MM + m] - sq;
                xp[fr * MM + m] = v;
                mx = fmaxf(mx, v);
            }
            pmx[fr * 4 + fq] = mx;
        }
        __syncthreads();
        // Phase C: exp + eps, write k'^T (transposed store for kv GEMM A).
        {
            float mx = fmaxf(fmaxf(pmx[fr * 4 + 0], pmx[fr * 4 + 1]),
                             fmaxf(pmx[fr * 4 + 2], pmx[fr * 4 + 3]));
            for (int m = fq * 64; m < fq * 64 + 64; ++m) {
                float e = (__expf(xp[fr * MM + m] - mx) + 1e-6f) * INV_SQRT_M;
                kpt[m * 64 + fr] = f2u(e);
            }
        }
        __syncthreads();
        // Load v^T into (now dead) xp region; accumulate k_sum from kpt rows
        for (int i = tid; i < 512; i += 256) {
            int n = i >> 3, d8 = (i & 7) * 8;
            uint4 tq = *(const uint4*)(vg + (size_t)(n0 + n) * DD + d8);
            const unsigned short* s = (const unsigned short*)&tq;
#pragma unroll
            for (int t = 0; t < 8; ++t) vt[(d8 + t) * 64 + n] = s[t];
        }
        {
            float s = 0.0f;
            for (int n = 0; n < 64; ++n) s += u2f(kpt[tid * 64 + n]);
            ksacc += s;
        }
        __syncthreads();
        // kv += k'^T v : M=256, N=64, K=64
#pragma unroll
        for (int ks = 0; ks < 64; ks += 32) {
            bf16x16 a0 = load_frag_a(kpt, 64, wave * 32, ks, lane);
            bf16x16 a1 = load_frag_a(kpt, 64, wave * 32 + 16, ks, lane);
#pragma unroll
            for (int j = 0; j < 4; ++j) {
                bf16x16 b = load_frag_b(vt, 64, ks, j * 16, lane);
                kvacc[0][j] = wmma_bf16(a0, b, kvacc[0][j]);
                kvacc[1][j] = wmma_bf16(a1, b, kvacc[1][j]);
            }
        }
    }
    // Write kv (bf16 [m][d]) and k_sum
#pragma unroll
    for (int i = 0; i < 2; ++i)
#pragma unroll
        for (int j = 0; j < 4; ++j)
#pragma unroll
            for (int r = 0; r < 8; ++r) {
                int m  = wave * 32 + i * 16 + ((lane >> 4) & 1) * 8 + r;
                int dd = j * 16 + (lane & 15);
                kvb[(size_t)bh * MM * DD + m * DD + dd] = f2u(kvacc[i][j][r]);
            }
    ksum_g[bh * MM + tid] = ksacc;
}

// ===========================================================================
// Kernel 3: per (b,h,chunk): q' = featmap(q); num = q' @ kv; den = q'.k_sum;
// write head-merged bf16 outh[(b,n),(h*64+d)] = num/den.
// ===========================================================================
__global__ __launch_bounds__(256) void num_kernel(const unsigned short* __restrict__ qhg,
                                                  const unsigned short* __restrict__ kvb,
                                                  const float* __restrict__ ksum_g,
                                                  const float* __restrict__ proj,
                                                  unsigned short* __restrict__ outh) {
    // LDS carve (~139 KB): projB | qb | qp | ks | denom | (xp ovl kvT) | partials
    __shared__ __align__(16) unsigned char smem[142592];
    unsigned short* projB = (unsigned short*)(smem);           // [256][64]
    unsigned short* qb    = (unsigned short*)(smem + 32768);   // [64][64]
    unsigned short* qp    = (unsigned short*)(smem + 40960);   // [64][256] q' (A, K=m)
    float*          ks    = (float*)(smem + 73728);            // [256]
    float*          denom = (float*)(smem + 74752);            // [64]
    float*          xp    = (float*)(smem + 75008);            // [64][256]
    unsigned short* kvT   = (unsigned short*)(smem + 75008);   // [64][256] kv^T (ovl xp)
    float*          psq   = (float*)(smem + 140544);           // [64][4] sq / den partials
    float*          pmx   = (float*)(smem + 141568);           // [64][4] max partials

    const int tid = threadIdx.x, lane = tid & 31, wave = tid >> 5;
    const int chunk = blockIdx.x;       // 0..63 -> n0
    const int bh = blockIdx.y;          // 0..63
    const int h = bh & (HH - 1), b = bh >> 4;
    const int n0 = chunk * 64;
    const int fr = tid & 63, fq = tid >> 6;

    for (int i = tid; i < MM * DD / 4; i += 256) {
        const float4 f = ((const float4*)(proj + (size_t)h * MM * DD))[i];
        uint2 t;
        t.x = pack2(f.x, f.y);
        t.y = pack2(f.z, f.w);
        ((uint2*)projB)[i] = t;
    }
    for (int i = tid; i < 512; i += 256)
        ((uint4*)qb)[i] = *(const uint4*)(qhg + (size_t)bh * NN * DD + (size_t)n0 * DD + i * 8);
    if (tid < MM) ks[tid] = ksum_g[bh * MM + tid];
    __builtin_prefetch(kvb + (size_t)bh * MM * DD + (size_t)tid * 64, 0, 0);
    __syncthreads();

    // xproj = (q * DN) @ proj^T
    {
        floatx8 xacc[8] = {};
        const int mt = (wave & 3) * 16;
        const int nb = (wave >> 2) * 128;
#pragma unroll
        for (int kk = 0; kk < 64; kk += 32) {
            bf16x16 a = load_frag_a(qb, 64, mt, kk, lane);
#pragma unroll
            for (int j = 0; j < 8; ++j) {
                bf16x16 bfr = load_frag_b(projB, 64, kk, nb + j * 16, lane);
                xacc[j] = wmma_bf16(a, bfr, xacc[j]);
            }
        }
#pragma unroll
        for (int j = 0; j < 8; ++j)
#pragma unroll
            for (int r = 0; r < 8; ++r) {
                int row = mt + ((lane >> 4) & 1) * 8 + r;
                int col = nb + j * 16 + (lane & 15);
                xp[row * MM + col] = xacc[j][r] * DATA_NORM;
            }
    }
    __syncthreads();
    // Feature map, 4 threads/row. Phase A: sq-norm partials.
    {
        float s = 0.0f;
        for (int dd = fq * 16; dd < fq * 16 + 16; ++dd) {
            float v = u2f(qb[fr * DD + dd]) * DATA_NORM;
            s += v * v;
        }
        psq[fr * 4 + fq] = s;
    }
    __syncthreads();
    // Phase B: subtract sq, rowwise-max partials.
    {
        float sq = 0.5f * (psq[fr * 4 + 0] + psq[fr * 4 + 1] +
                           psq[fr * 4 + 2] + psq[fr * 4 + 3]);
        float mx = -1e30f;
        for (int m = fq * 64; m < fq * 64 + 64; ++m) {
            float v = xp[fr * MM + m] - sq;
            xp[fr * MM + m] = v;
            mx = fmaxf(mx, v);
        }
        pmx[fr * 4 + fq] = mx;
    }
    __syncthreads();
    // Phase C: exp + eps -> q' (A of numerator GEMM); denominator partials.
    {
        float mx = fmaxf(fmaxf(pmx[fr * 4 + 0], pmx[fr * 4 + 1]),
                         fmaxf(pmx[fr * 4 + 2], pmx[fr * 4 + 3]));
        float den = 0.0f;
        for (int m = fq * 64; m < fq * 64 + 64; ++m) {
            float e = (__expf(xp[fr * MM + m] - mx) + 1e-6f) * INV_SQRT_M;
            qp[fr * MM + m] = f2u(e);
            den += e * ks[m];
        }
        psq[fr * 4 + fq] = den;  // reuse sq partial buffer for den partials
    }
    __syncthreads();
    // Load kv^T into (now dead) xp region; finish denominator.
    for (int i = tid; i < 2048; i += 256) {
        int m = i >> 3, d8 = (i & 7) * 8;
        uint4 tq = *(const uint4*)(kvb + (size_t)bh * MM * DD + m * DD + d8);
        const unsigned short* s = (const unsigned short*)&tq;
#pragma unroll
        for (int t = 0; t < 8; ++t) kvT[(d8 + t) * MM + m] = s[t];
    }
    if (tid < 64)
        denom[tid] = psq[tid * 4 + 0] + psq[tid * 4 + 1] + psq[tid * 4 + 2] +
                     psq[tid * 4 + 3] + 1e-6f;
    __syncthreads();
    // num = q' @ kv : M=64, N=64, K=256. Wave -> mt=(w&3), nt base=(w>>2)*2.
    floatx8 nacc[2] = {};
    const int mt = (wave & 3) * 16;
    const int ntb = (wave >> 2) * 2;
#pragma unroll
    for (int k0 = 0; k0 < MM; k0 += 32) {
        bf16x16 a = load_frag_a(qp, MM, mt, k0, lane);
#pragma unroll
        for (int j = 0; j < 2; ++j) {
            bf16x16 bfr = load_frag_b(kvT, MM, k0, (ntb + j) * 16, lane);
            nacc[j] = wmma_bf16(a, bfr, nacc[j]);
        }
    }
#pragma unroll
    for (int j = 0; j < 2; ++j)
#pragma unroll
        for (int r = 0; r < 8; ++r) {
            int row = mt + ((lane >> 4) & 1) * 8 + r;
            int col = (ntb + j) * 16 + (lane & 15);
            float val = nacc[j][r] / denom[row];
            int n = n0 + row;
            outh[((size_t)(b * NN + n)) * CC + h * DD + col] = f2u(val);
        }
}

// ===========================================================================
// Kernel 4: y = outh @ W_out + b_out  (f32 output)
// ===========================================================================
__global__ __launch_bounds__(256) void out_gemm(const unsigned short* __restrict__ a_bf16,
                                                const float* __restrict__ w,
                                                const float* __restrict__ bias,
                                                float* __restrict__ y) {
    __shared__ __align__(16) unsigned short As[128 * 64];
    __shared__ __align__(16) unsigned short Bt[128 * 64];
    const int tid = threadIdx.x, lane = tid & 31, wave = tid >> 5;
    const int m_base = blockIdx.y * 128;
    const int n_base = blockIdx.x * 128;
    const int wm = (wave & 3) * 32;
    const int wn = (wave >> 2) * 64;

    floatx8 acc[2][4] = {};

    for (int k0 = 0; k0 < CC; k0 += 64) {
        __syncthreads();
        // A stage: source already bf16 -> straight b128 copies
        for (int i = tid; i < 1024; i += 256) {
            int r = i >> 3, c8 = (i & 7) * 8;
            ((uint4*)As)[i] = *(const uint4*)(a_bf16 + (size_t)(m_base + r) * CC + k0 + c8);
        }
        // B stage: coalesced float4 along N, transposed bf16 scatter
        for (int i = tid; i < 2048; i += 256) {
            int kk = i >> 5, n = (i & 31) * 4;
            const float4 f = *(const float4*)(w + (size_t)(k0 + kk) * CC + n_base + n);
            Bt[(n + 0) * 64 + kk] = f2u(f.x);
            Bt[(n + 1) * 64 + kk] = f2u(f.y);
            Bt[(n + 2) * 64 + kk] = f2u(f.z);
            Bt[(n + 3) * 64 + kk] = f2u(f.w);
        }
        if (k0 + 64 < CC) {
            __builtin_prefetch(a_bf16 + (size_t)(m_base + (tid >> 1)) * CC + k0 + 64 + (tid & 1) * 32, 0, 0);
            __builtin_prefetch(w + (size_t)(k0 + 64 + (tid >> 2)) * CC + n_base + (tid & 3) * 32, 0, 0);
        }
        __syncthreads();
#pragma unroll
        for (int kk = 0; kk < 64; kk += 32) {
            bf16x16 a0 = load_frag_a(As, 64, wm, kk, lane);
            bf16x16 a1 = load_frag_a(As, 64, wm + 16, kk, lane);
#pragma unroll
            for (int j = 0; j < 4; ++j) {
                bf16x16 b = load_frag_b(Bt, 64, kk, wn + j * 16, lane);
                acc[0][j] = wmma_bf16(a0, b, acc[0][j]);
                acc[1][j] = wmma_bf16(a1, b, acc[1][j]);
            }
        }
    }
#pragma unroll
    for (int i = 0; i < 2; ++i)
#pragma unroll
        for (int j = 0; j < 4; ++j)
#pragma unroll
            for (int r = 0; r < 8; ++r) {
                int row = m_base + wm + i * 16 + ((lane >> 4) & 1) * 8 + r;
                int col = n_base + wn + j * 16 + (lane & 15);
                y[(size_t)row * CC + col] = acc[i][j][r] + bias[col];
            }
}

// ===========================================================================
extern "C" void kernel_launch(void* const* d_in, const int* in_sizes, int n_in,
                              void* d_out, int out_size, void* d_ws, size_t ws_size,
                              hipStream_t stream) {
    const float* x     = (const float*)d_in[0];
    const float* w_qkv = (const float*)d_in[1];
    const float* b_qkv = (const float*)d_in[2];
    const float* w_out = (const float*)d_in[3];
    const float* b_out = (const float*)d_in[4];
    const float* proj  = (const float*)d_in[5];

    const size_t HEAD_ELEMS = (size_t)BB * HH * NN * DD;  // 16,777,216
    const size_t KV_ELEMS   = (size_t)BB * HH * MM * DD;  // 1,048,576
    const size_t OUT_ELEMS  = (size_t)BB * NN * CC;       // 16,777,216

    unsigned short* qh   = (unsigned short*)d_ws;
    unsigned short* kh   = qh + HEAD_ELEMS;
    unsigned short* vh   = kh + HEAD_ELEMS;
    unsigned short* kvb  = vh + HEAD_ELEMS;
    unsigned short* outh = kvb + KV_ELEMS;
    float*          ksum = (float*)(outh + OUT_ELEMS);    // 16,384 f32

    // 1) qkv projection GEMM (16384x3072x1024)
    qkv_gemm<<<dim3(3 * CC / 128, (BB * NN) / 128), 256, 0, stream>>>(
        x, w_qkv, b_qkv, qh, kh, vh);
    // 2) k feature map + kv/k_sum accumulation (one WG per (b,h))
    kv_kernel<<<dim3(BB * HH), 256, 0, stream>>>(kh, vh, proj, kvb, ksum);
    // 3) q feature map + numerator/denominator (WG per (b,h,64-row chunk))
    num_kernel<<<dim3(NN / 64, BB * HH), 256, 0, stream>>>(qh, kvb, ksum, proj, outh);
    // 4) output projection GEMM (16384x1024x1024)
    out_gemm<<<dim3(CC / 128, (BB * NN) / 128), 256, 0, stream>>>(
        outh, w_out, b_out, (float*)d_out);
}
